// AdaptiveFourierNeuralOperator_67972152426697
// MI455X (gfx1250) — matrix operand order repaired and checked by reading, not compiled, not run on hardware
//
#include <hip/hip_runtime.h>

// ---------------------------------------------------------------------------
// AFNO (Adaptive Fourier Neural Operator) for MI455X / gfx1250, all-WMMA.
// Every stage is a GEMM on v_wmma_f32_16x16x32_f16 (f16 in, f32 accumulate).
// Bias GEMM uses double-buffered GLOBAL_LOAD_ASYNC_TO_LDS_B128 (ASYNCcnt).
//   dims: B=8, H=64, W=64, C=768, nb=4, bs=192, Wf=48 (padded rfft width)
// ---------------------------------------------------------------------------

typedef __attribute__((ext_vector_type(16))) _Float16 v16h;
typedef __attribute__((ext_vector_type(8)))  _Float16 v8h;
typedef __attribute__((ext_vector_type(8)))  float    v8f;

#define WMMA_F16(a, b, c) \
  __builtin_amdgcn_wmma_f32_16x16x32_f16(false, (a), false, (b), (short)0, (c), false, false)

__device__ __forceinline__ v8f vzero8() {
  v8f z;
  #pragma unroll
  for (int i = 0; i < 8; ++i) z[i] = 0.0f;
  return z;
}

// A-fragment, 16x32 f16, from row-major [m][k] LDS tile, row stride ld halves
// (ld multiple of 8, rows 16B aligned). Lane l: row m0+(l&15),
// K = k0 + (l>=16 ? 16 : 0) + e.
__device__ __forceinline__ v16h load_a(const _Float16* A, int ld, int m0, int k0) {
  const int l = threadIdx.x & 31;
  const _Float16* p = A + (m0 + (l & 15)) * ld + k0 + ((l >> 4) << 4);
  v8h lo = *(const v8h*)(p);
  v8h hi = *(const v8h*)(p + 8);
  return __builtin_shufflevector(lo, hi, 0, 1, 2, 3, 4, 5, 6, 7,
                                 8, 9, 10, 11, 12, 13, 14, 15);
}

// B-fragment, 32x16 f16, from [n][k] LDS tile, stride ld halves (multiple of
// 16, rows 32B aligned). Lane l: col n0+(l&15), K = k0 + (l>=16 ? 16 : 0) + e.
__device__ __forceinline__ v16h load_b(const _Float16* B, int ld, int n0, int k0) {
  const int l = threadIdx.x & 31;
  const _Float16* p = B + (n0 + (l & 15)) * ld + k0 + ((l >> 4) << 4);
  return *(const v16h*)(p);
}

// f32-in-LDS variants: fragment conversion at load time (v_cvt co-executes
// with WMMA, which is tracked as a TRANS op on CDNA5).
__device__ __forceinline__ v16h load_a32(const float* A, int ld, int m0, int k0) {
  const int l = threadIdx.x & 31;
  const float* p = A + (m0 + (l & 15)) * ld + k0 + ((l >> 4) << 4);
  v8f lo = *(const v8f*)(p);
  v8f hi = *(const v8f*)(p + 8);
  v16h r;
  #pragma unroll
  for (int i = 0; i < 8; ++i) { r[i] = (_Float16)lo[i]; r[i + 8] = (_Float16)hi[i]; }
  return r;
}
__device__ __forceinline__ v16h load_b32(const float* B, int ld, int n0, int k0) {
  const int l = threadIdx.x & 31;
  const float* p = B + (n0 + (l & 15)) * ld + k0 + ((l >> 4) << 4);
  v8f lo = *(const v8f*)(p);
  v8f hi = *(const v8f*)(p + 8);
  v16h r;
  #pragma unroll
  for (int i = 0; i < 8; ++i) { r[i] = (_Float16)lo[i]; r[i + 8] = (_Float16)hi[i]; }
  return r;
}

// ---- CDNA5 async global -> LDS copy (VGLOBAL async encoding, ASYNCcnt) ----
// VDST = per-lane LDS byte offset (generic LDS pointer truncates to the LDS
// offset: aperture base lives in addr[63:32]); VADDR = 64-bit global address.
__device__ __forceinline__ void async_g2l_b128(const float* lds_dst, const float* gsrc) {
  unsigned int loff = (unsigned int)(unsigned long long)lds_dst;
  asm volatile("global_load_async_to_lds_b128 %0, %1, off"
               :: "v"(loff), "v"(gsrc) : "memory");
}
__device__ __forceinline__ void wait_asynccnt8() {
  asm volatile("s_wait_asynccnt 0x8" ::: "memory");
}
__device__ __forceinline__ void wait_asynccnt0() {
  asm volatile("s_wait_asynccnt 0x0" ::: "memory");
}

// Copy one 128x32 f32 tile (global row stride 768) into LDS (row stride 40
// floats). 256 threads x 4 iters x 16B = 16KB; 4 async ops per wave.
__device__ __forceinline__ void stage_tile_async(float (*lds)[40],
                                                 const float* g, int tid) {
  #pragma unroll
  for (int it = 0; it < 4; ++it) {
    int idx = tid + it * 256;        // 16B segment index, 8 segs per row
    int r = idx >> 3, seg = (idx & 7) << 2;
    async_g2l_b128(&lds[r][seg], g + r * 768 + seg);
  }
}

// ---------------------------------------------------------------------------
// Stage 1: bias path  out[t][n] = sum_k x[t][k] * Wb[n][k] + bb[n]
// M = 32768 tokens, N = K = 768.  128x128 tile / workgroup.
// Double-buffered async-LDS pipeline over 24 K-steps of 32.
// ---------------------------------------------------------------------------
__global__ __launch_bounds__(256) void k_bias_gemm(const float* __restrict__ x,
                                                   const float* __restrict__ Wb,
                                                   const float* __restrict__ bb,
                                                   float* __restrict__ out) {
  __shared__ __align__(32) float As32[2][128][40];
  __shared__ __align__(32) float Bs32[2][128][40];
  const int c0  = blockIdx.x * 128;
  const int t0  = blockIdx.y * 128;
  const int tid = threadIdx.x;
  const int wid = tid >> 5, l = tid & 31;
  const int m0w = (wid & 1) * 64;   // wave's 64-row half  (4 m-tiles)
  const int n0w = (wid >> 1) * 32;  // wave's 32-col strip (2 n-tiles)

  v8f acc[4][2];
  #pragma unroll
  for (int i = 0; i < 4; ++i)
    for (int j = 0; j < 2; ++j) acc[i][j] = vzero8();

  // preload K-tile 0 (8 async ops / wave)
  stage_tile_async(As32[0], x + t0 * 768, tid);
  stage_tile_async(Bs32[0], Wb + c0 * 768, tid);

  for (int ks = 0; ks < 24; ++ks) {
    __syncthreads();  // everyone done reading buf[(ks+1)&1] from iter ks-1
    bool pf = (ks + 1) < 24;
    if (pf) {  // uniform over the block
      int kn = (ks + 1) * 32;
      stage_tile_async(As32[(ks + 1) & 1], x + t0 * 768 + kn, tid);
      stage_tile_async(Bs32[(ks + 1) & 1], Wb + c0 * 768 + kn, tid);
    }
    // async loads retire in order: <=8 outstanding => current tile landed
    if (pf) wait_asynccnt8(); else wait_asynccnt0();
    __syncthreads();  // whole tile visible to all waves

    const float* Ab = &As32[ks & 1][0][0];
    const float* Bb = &Bs32[ks & 1][0][0];
    v16h bf0 = load_b32(Bb, 40, n0w, 0);
    v16h bf1 = load_b32(Bb, 40, n0w + 16, 0);
    #pragma unroll
    for (int i = 0; i < 4; ++i) {
      v16h af = load_a32(Ab, 40, m0w + 16 * i, 0);
      acc[i][0] = WMMA_F16(af, bf0, acc[i][0]);
      acc[i][1] = WMMA_F16(af, bf1, acc[i][1]);
    }
  }
  #pragma unroll
  for (int j = 0; j < 2; ++j) {
    int n = c0 + n0w + 16 * j + (l & 15);
    float bias = bb[n];
    #pragma unroll
    for (int i = 0; i < 4; ++i)
      #pragma unroll
      for (int r = 0; r < 8; ++r) {
        int m = t0 + m0w + 16 * i + r + 8 * (l >> 4);
        out[m * 768 + n] = acc[i][j][r] + bias;
      }
  }
}

// ---------------------------------------------------------------------------
// Stage 2: rDFT along W.  Per (b,h):  G[w'][c] = sum_w T[w'][w] * x[w][c]
// Twiddles (cos/64, -sin/64) generated into LDS.  M=48(w'), K=64(w), N=128(c).
// ---------------------------------------------------------------------------
__global__ __launch_bounds__(256) void k_dft_w(const float* __restrict__ x,
                                               float* __restrict__ Gr,
                                               float* __restrict__ Gi) {
  __shared__ __align__(32) _Float16 Cw[48][80];
  __shared__ __align__(32) _Float16 Sw[48][80];
  __shared__ __align__(32) _Float16 Bs[128][80];  // [c][w]
  const int c0  = blockIdx.x * 128;
  const int bh  = blockIdx.y;  // b*64 + h
  const int tid = threadIdx.x;
  const int wid = tid >> 5, l = tid & 31;

  for (int idx = tid; idx < 48 * 64; idx += 256) {
    int m = idx >> 6, k = idx & 63;
    float ang = (float)((m * k) & 63) * 0.09817477042468103f;  // 2*pi/64
    float s, c;
    __sincosf(ang, &s, &c);
    Cw[m][k] = (_Float16)(c * 0.015625f);    // cos / 64 (ortho fwd scale)
    Sw[m][k] = (_Float16)(-s * 0.015625f);   // -sin / 64
  }
  for (int idx = tid; idx < 64 * 128; idx += 256) {
    int w = idx >> 7, c = idx & 127;
    Bs[c][w] = (_Float16)x[(bh * 64 + w) * 768 + c0 + c];
  }
  __syncthreads();

  const int n0w = wid * 16;
  v8f ar[3], ai[3];
  #pragma unroll
  for (int m = 0; m < 3; ++m) { ar[m] = vzero8(); ai[m] = vzero8(); }

  #pragma unroll
  for (int k0 = 0; k0 < 64; k0 += 32) {
    v16h bf = load_b(&Bs[0][0], 80, n0w, k0);
    #pragma unroll
    for (int mt = 0; mt < 3; ++mt) {
      v16h ac = load_a(&Cw[0][0], 80, mt * 16, k0);
      v16h as = load_a(&Sw[0][0], 80, mt * 16, k0);
      ar[mt] = WMMA_F16(ac, bf, ar[mt]);
      ai[mt] = WMMA_F16(as, bf, ai[mt]);
    }
  }
  #pragma unroll
  for (int mt = 0; mt < 3; ++mt)
    #pragma unroll
    for (int r = 0; r < 8; ++r) {
      int wp = mt * 16 + r + 8 * (l >> 4);
      int n  = c0 + n0w + (l & 15);
      int o  = (bh * 48 + wp) * 768 + n;
      Gr[o] = ar[mt][r];
      Gi[o] = ai[mt][r];
    }
}

// ---------------------------------------------------------------------------
// Stage 3 / 5: complex DFT along H (sgn=+1 forward e^{-i}, sgn=-1 inverse).
//   Fr = C*gr + sgn*S*gi ;  Fi = C*gi - sgn*S*gr     (per (b, w'<33))
// ---------------------------------------------------------------------------
__global__ __launch_bounds__(256) void k_dft_h(const float* __restrict__ Inr,
                                               const float* __restrict__ Ini,
                                               float* __restrict__ Outr,
                                               float* __restrict__ Outi,
                                               float sgn) {
  __shared__ __align__(32) _Float16 Ch[64][80];
  __shared__ __align__(32) _Float16 Sp[64][80];   // +sgn*sin
  __shared__ __align__(32) _Float16 Sn[64][80];   // -sgn*sin
  __shared__ __align__(32) _Float16 Brs[128][80]; // [c][h]
  __shared__ __align__(32) _Float16 Bis[128][80];
  const int c0  = blockIdx.x * 128;
  const int b   = blockIdx.y / 33;
  const int wp  = blockIdx.y % 33;
  const int tid = threadIdx.x;
  const int wid = tid >> 5, l = tid & 31;

  for (int idx = tid; idx < 64 * 64; idx += 256) {
    int m = idx >> 6, k = idx & 63;
    float ang = (float)((m * k) & 63) * 0.09817477042468103f;
    float s, c;
    __sincosf(ang, &s, &c);
    Ch[m][k] = (_Float16)c;
    Sp[m][k] = (_Float16)(sgn * s);
    Sn[m][k] = (_Float16)(-sgn * s);
  }
  for (int idx = tid; idx < 64 * 128; idx += 256) {
    int h = idx >> 7, c = idx & 127;
    int o = ((b * 64 + h) * 48 + wp) * 768 + c0 + c;
    Brs[c][h] = (_Float16)Inr[o];
    Bis[c][h] = (_Float16)Ini[o];
  }
  __syncthreads();

  const int n0w = wid * 16;
  v8f fr[4], fi[4];
  #pragma unroll
  for (int m = 0; m < 4; ++m) { fr[m] = vzero8(); fi[m] = vzero8(); }

  #pragma unroll
  for (int k0 = 0; k0 < 64; k0 += 32) {
    v16h br = load_b(&Brs[0][0], 80, n0w, k0);
    v16h bi = load_b(&Bis[0][0], 80, n0w, k0);
    #pragma unroll
    for (int mt = 0; mt < 4; ++mt) {
      v16h ac = load_a(&Ch[0][0], 80, mt * 16, k0);
      v16h as = load_a(&Sp[0][0], 80, mt * 16, k0);
      v16h an = load_a(&Sn[0][0], 80, mt * 16, k0);
      fr[mt] = WMMA_F16(ac, br, fr[mt]);
      fr[mt] = WMMA_F16(as, bi, fr[mt]);
      fi[mt] = WMMA_F16(ac, bi, fi[mt]);
      fi[mt] = WMMA_F16(an, br, fi[mt]);
    }
  }
  #pragma unroll
  for (int mt = 0; mt < 4; ++mt)
    #pragma unroll
    for (int r = 0; r < 8; ++r) {
      int hp = mt * 16 + r + 8 * (l >> 4);
      int o  = ((b * 64 + hp) * 48 + wp) * 768 + c0 + n0w + (l & 15);
      Outr[o] = fr[mt][r];
      Outi[o] = fi[mt][r];
    }
}

// ---------------------------------------------------------------------------
// Stage 4: block-diagonal complex MLP in frequency domain.
// Per workgroup: 32 tokens x one block j (bs=192).  K staged by 32.
//   r1 = relu(xr*W1r - xi*W1i + b1r); i1 = relu(xr*W1i + xi*W1r + b1i)
//   r2 = r1*W2r - i1*W2i + b2r;       i2 = r2*W2i + i1*W2r + b2i   (new r2!)
// ---------------------------------------------------------------------------
__global__ __launch_bounds__(256) void k_mlp(const float* __restrict__ Fr,
                                             const float* __restrict__ Fi,
                                             const float* __restrict__ w1,
                                             const float* __restrict__ b1,
                                             const float* __restrict__ w2,
                                             const float* __restrict__ b2,
                                             float* __restrict__ Mr,
                                             float* __restrict__ Mi) {
  __shared__ __align__(32) _Float16 xr_s[32][208];
  __shared__ __align__(32) _Float16 xi_s[32][208];
  __shared__ __align__(32) _Float16 Wr_s[192][48];  // [n][dd] staged K-slice
  __shared__ __align__(32) _Float16 Wi_s[192][48];
  __shared__ __align__(32) _Float16 r1_s[32][208];
  __shared__ __align__(32) _Float16 i1_s[32][208];
  __shared__ __align__(32) _Float16 r2_s[32][208];
  const int t0  = blockIdx.x * 32;
  const int j   = blockIdx.y;  // block 0..3
  const int tid = threadIdx.x;
  const int wid = tid >> 5, l = tid & 31;
  const int m0w = (wid & 1) * 16;  // 2 m-tiles
  const int ng  = wid >> 1;        // 4 n-groups of 3 n-tiles

  const float* W1r = w1 + (0 * 4 + j) * 36864;
  const float* W1i = w1 + (1 * 4 + j) * 36864;
  const float* W2r = w2 + (0 * 4 + j) * 36864;
  const float* W2i = w2 + (1 * 4 + j) * 36864;

  // warm L2 with layer-2 weights while layer 1 runs (global_prefetch)
  __builtin_prefetch(W2r + (tid << 6), 0, 1);
  __builtin_prefetch(W2i + (tid << 6), 0, 1);

  for (int idx = tid; idx < 32 * 192; idx += 256) {
    int r = idx / 192, c = idx % 192;
    int t = t0 + r, bh = t / 33, wp = t % 33;
    int o = (bh * 48 + wp) * 768 + j * 192 + c;
    xr_s[r][c] = (_Float16)Fr[o];
    xi_s[r][c] = (_Float16)Fi[o];
  }

  // ---- layer 1 ----
  v8f a_rr[3], a_ii[3], a_ri[3], a_ir[3];
  #pragma unroll
  for (int q = 0; q < 3; ++q) {
    a_rr[q] = vzero8(); a_ii[q] = vzero8();
    a_ri[q] = vzero8(); a_ir[q] = vzero8();
  }
  for (int ks = 0; ks < 6; ++ks) {
    int k0 = ks * 32;
    __syncthreads();
    for (int idx = tid; idx < 192 * 32; idx += 256) {
      int n = idx % 192, dd = idx / 192;  // global reads contiguous in n
      Wr_s[n][dd] = (_Float16)W1r[(k0 + dd) * 192 + n];
      Wi_s[n][dd] = (_Float16)W1i[(k0 + dd) * 192 + n];
    }
    __syncthreads();
    v16h ar = load_a(&xr_s[0][0], 208, m0w, k0);
    v16h ai = load_a(&xi_s[0][0], 208, m0w, k0);
    #pragma unroll
    for (int q = 0; q < 3; ++q) {
      int n0 = ng * 48 + q * 16;
      v16h br = load_b(&Wr_s[0][0], 48, n0, 0);
      v16h bi = load_b(&Wi_s[0][0], 48, n0, 0);
      a_rr[q] = WMMA_F16(ar, br, a_rr[q]);
      a_ii[q] = WMMA_F16(ai, bi, a_ii[q]);
      a_ri[q] = WMMA_F16(ar, bi, a_ri[q]);
      a_ir[q] = WMMA_F16(ai, br, a_ir[q]);
    }
  }
  #pragma unroll
  for (int q = 0; q < 3; ++q) {
    int ncol = ng * 48 + q * 16 + (l & 15);
    float b1r = b1[(0 * 4 + j) * 192 + ncol];
    float b1i = b1[(1 * 4 + j) * 192 + ncol];
    #pragma unroll
    for (int r = 0; r < 8; ++r) {
      int row = m0w + r + 8 * (l >> 4);
      r1_s[row][ncol] = (_Float16)fmaxf(a_rr[q][r] - a_ii[q][r] + b1r, 0.0f);
      i1_s[row][ncol] = (_Float16)fmaxf(a_ri[q][r] + a_ir[q][r] + b1i, 0.0f);
    }
  }
  __syncthreads();

  // ---- layer 2a: r2 ----
  v8f c_a[3], c_b[3];
  #pragma unroll
  for (int q = 0; q < 3; ++q) { c_a[q] = vzero8(); c_b[q] = vzero8(); }
  for (int ks = 0; ks < 6; ++ks) {
    int k0 = ks * 32;
    __syncthreads();
    for (int idx = tid; idx < 192 * 32; idx += 256) {
      int n = idx % 192, dd = idx / 192;
      Wr_s[n][dd] = (_Float16)W2r[(k0 + dd) * 192 + n];
      Wi_s[n][dd] = (_Float16)W2i[(k0 + dd) * 192 + n];
    }
    __syncthreads();
    v16h ar = load_a(&r1_s[0][0], 208, m0w, k0);
    v16h ai = load_a(&i1_s[0][0], 208, m0w, k0);
    #pragma unroll
    for (int q = 0; q < 3; ++q) {
      int n0 = ng * 48 + q * 16;
      v16h br = load_b(&Wr_s[0][0], 48, n0, 0);
      v16h bi = load_b(&Wi_s[0][0], 48, n0, 0);
      c_a[q] = WMMA_F16(ar, br, c_a[q]);
      c_b[q] = WMMA_F16(ai, bi, c_b[q]);
    }
  }
  #pragma unroll
  for (int q = 0; q < 3; ++q) {
    int ncol = ng * 48 + q * 16 + (l & 15);
    float b2r = b2[(0 * 4 + j) * 192 + ncol];
    #pragma unroll
    for (int r = 0; r < 8; ++r) {
      int row = m0w + r + 8 * (l >> 4);
      float rv = c_a[q][r] - c_b[q][r] + b2r;
      r2_s[row][ncol] = (_Float16)rv;
      int t = t0 + row, bh = t / 33, wp = t % 33;
      Mr[(bh * 48 + wp) * 768 + j * 192 + ncol] = rv;
    }
  }
  __syncthreads();

  // ---- layer 2b: i2 = r2*W2i + i1*W2r + b2i (W2 slice restaged per k) ----
  v8f c_c[3], c_d[3];
  #pragma unroll
  for (int q = 0; q < 3; ++q) { c_c[q] = vzero8(); c_d[q] = vzero8(); }
  for (int ks = 0; ks < 6; ++ks) {
    int k0 = ks * 32;
    __syncthreads();
    for (int idx = tid; idx < 192 * 32; idx += 256) {
      int n = idx % 192, dd = idx / 192;
      Wr_s[n][dd] = (_Float16)W2r[(k0 + dd) * 192 + n];
      Wi_s[n][dd] = (_Float16)W2i[(k0 + dd) * 192 + n];
    }
    __syncthreads();
    v16h a2 = load_a(&r2_s[0][0], 208, m0w, k0);
    v16h ai = load_a(&i1_s[0][0], 208, m0w, k0);
    #pragma unroll
    for (int q = 0; q < 3; ++q) {
      int n0 = ng * 48 + q * 16;
      v16h br = load_b(&Wr_s[0][0], 48, n0, 0);
      v16h bi = load_b(&Wi_s[0][0], 48, n0, 0);
      c_c[q] = WMMA_F16(a2, bi, c_c[q]);
      c_d[q] = WMMA_F16(ai, br, c_d[q]);
    }
  }
  #pragma unroll
  for (int q = 0; q < 3; ++q) {
    int ncol = ng * 48 + q * 16 + (l & 15);
    float b2i = b2[(1 * 4 + j) * 192 + ncol];
    #pragma unroll
    for (int r = 0; r < 8; ++r) {
      int row = m0w + r + 8 * (l >> 4);
      int t = t0 + row, bh = t / 33, wp = t % 33;
      Mi[(bh * 48 + wp) * 768 + j * 192 + ncol] = c_c[q][r] + c_d[q][r] + b2i;
    }
  }
}

// ---------------------------------------------------------------------------
// Stage 6: irfft along W (hermitian-folded) + residual add of bias (in d_out).
//   out[w][c] += sum_{w'} alpha*cos/64 * Fr - alpha*sin/64 * Fi
//   alpha = 1 (w'=0,32), 2 (1..31), 0 (>=33);  LDS also zero-fills w'>=33.
// ---------------------------------------------------------------------------
__global__ __launch_bounds__(256) void k_idft_w(const float* __restrict__ Fr,
                                                const float* __restrict__ Fi,
                                                float* __restrict__ out) {
  __shared__ __align__(32) _Float16 Ar[64][80];
  __shared__ __align__(32) _Float16 Ai[64][80];
  __shared__ __align__(32) _Float16 Brs[128][80];  // [c][w']
  __shared__ __align__(32) _Float16 Bis[128][80];
  const int c0  = blockIdx.x * 128;
  const int bh  = blockIdx.y;  // b*64 + h
  const int tid = threadIdx.x;
  const int wid = tid >> 5, l = tid & 31;

  for (int idx = tid; idx < 64 * 64; idx += 256) {
    int m = idx >> 6, k = idx & 63;  // m = w out, k = w' in
    float alpha = (k == 0 || k == 32) ? 1.0f : (k < 33 ? 2.0f : 0.0f);
    float ang = (float)((m * k) & 63) * 0.09817477042468103f;
    float s, c;
    __sincosf(ang, &s, &c);
    Ar[m][k] = (_Float16)(alpha * c * 0.015625f);
    Ai[m][k] = (_Float16)(-alpha * s * 0.015625f);
  }
  for (int idx = tid; idx < 64 * 128; idx += 256) {
    int wp = idx >> 7, c = idx & 127;
    float vr = 0.0f, vi = 0.0f;
    if (wp < 33) {
      int o = (bh * 48 + wp) * 768 + c0 + c;
      vr = Fr[o];
      vi = Fi[o];
    }
    Brs[c][wp] = (_Float16)vr;
    Bis[c][wp] = (_Float16)vi;
  }
  __syncthreads();

  const int n0w = wid * 16;
  v8f acc[4];
  #pragma unroll
  for (int m = 0; m < 4; ++m) acc[m] = vzero8();

  #pragma unroll
  for (int k0 = 0; k0 < 64; k0 += 32) {
    v16h br = load_b(&Brs[0][0], 80, n0w, k0);
    v16h bi = load_b(&Bis[0][0], 80, n0w, k0);
    #pragma unroll
    for (int mt = 0; mt < 4; ++mt) {
      v16h fr = load_a(&Ar[0][0], 80, mt * 16, k0);
      v16h fi = load_a(&Ai[0][0], 80, mt * 16, k0);
      acc[mt] = WMMA_F16(fr, br, acc[mt]);
      acc[mt] = WMMA_F16(fi, bi, acc[mt]);
    }
  }
  #pragma unroll
  for (int mt = 0; mt < 4; ++mt)
    #pragma unroll
    for (int r = 0; r < 8; ++r) {
      int w = mt * 16 + r + 8 * (l >> 4);
      int o = (bh * 64 + w) * 768 + c0 + n0w + (l & 15);
      out[o] = out[o] + acc[mt][r];  // residual add of Linear-bias path
    }
}

// ---------------------------------------------------------------------------
extern "C" void kernel_launch(void* const* d_in, const int* in_sizes, int n_in,
                              void* d_out, int out_size, void* d_ws, size_t ws_size,
                              hipStream_t stream) {
  (void)in_sizes; (void)n_in; (void)out_size; (void)ws_size;
  const float* x  = (const float*)d_in[0];
  const float* w1 = (const float*)d_in[1];
  const float* b1 = (const float*)d_in[2];
  const float* w2 = (const float*)d_in[3];
  const float* b2 = (const float*)d_in[4];
  const float* bw = (const float*)d_in[5];
  const float* bb = (const float*)d_in[6];
  float* out = (float*)d_out;
  float* ws  = (float*)d_ws;

  const size_t NF = (size_t)8 * 64 * 48 * 768;  // one freq-plane buffer (floats)
  float* Gr = ws;
  float* Gi = ws + NF;
  float* Fr = ws + 2 * NF;
  float* Fi = ws + 3 * NF;

  // 1) Linear bias path straight into d_out (async double-buffered GEMM)
  k_bias_gemm<<<dim3(6, 256), 256, 0, stream>>>(x, bw, bb, out);
  // 2) rDFT along W: x -> G
  k_dft_w<<<dim3(6, 512), 256, 0, stream>>>(x, Gr, Gi);
  // 3) forward complex DFT along H: G -> F (w' < 33 only)
  k_dft_h<<<dim3(6, 8 * 33), 256, 0, stream>>>(Gr, Gi, Fr, Fi, 1.0f);
  // 4) block-diagonal complex MLP: F -> G
  k_mlp<<<dim3(528, 4), 256, 0, stream>>>(Fr, Fi, w1, b1, w2, b2, Gr, Gi);
  // 5) inverse complex DFT along H: G -> F
  k_dft_h<<<dim3(6, 8 * 33), 256, 0, stream>>>(Gr, Gi, Fr, Fi, -1.0f);
  // 6) hermitian irfft along W + residual add into d_out
  k_idft_w<<<dim3(6, 512), 256, 0, stream>>>(Fr, Fi, out);
}
